// QuantizedAttention_65566970741151
// MI455X (gfx1250) — compile-verified, matrix operand order checked by reading, not run on hardware
//
#include <hip/hip_runtime.h>

// ---------------------------------------------------------------------------
// CDNA5 (gfx1250) quantized attention: bf16 WMMA pipeline, fp32 accumulate,
// async global->LDS staging (ASYNCcnt) with double buffering, NT streaming
// stores for the huge attn_weights output.
// ---------------------------------------------------------------------------

typedef __attribute__((ext_vector_type(16))) __bf16 v16bf;
typedef __attribute__((ext_vector_type(8)))  float  v8f;

#define EMBED   1024
#define HEADS   16
#define HDIM    64
#define SEQ     2048
#define BATCH   2
#define MROWS   (BATCH * SEQ)   // 4096

__device__ __forceinline__ __bf16 to_bf16(float f) {
    unsigned u = __builtin_bit_cast(unsigned, f);
    u += 0x7FFFu + ((u >> 16) & 1u);           // round-to-nearest-even
    unsigned short h = (unsigned short)(u >> 16);
    return __builtin_bit_cast(__bf16, h);
}

union FragU { v16bf v; uint4 q[2]; };

// Load one 16-lane x K=32 bf16 WMMA fragment (A or B operand) following the
// CDNA5 16-bit operand VGPR layout: lanes 0-15 hold K={0..7,16..23},
// lanes 16-31 hold K={8..15,24..31}.  p points at this lane's row (K index 0),
// hi = (lane&16) ? 8 : 0.  Two dword4 loads per lane.
__device__ __forceinline__ v16bf load_frag32(const __bf16* p, int hi) {
    FragU f;
    f.q[0] = *(const uint4*)(p + hi);
    f.q[1] = *(const uint4*)(p + 16 + hi);
    return f.v;
}

__device__ __forceinline__ v8f wmma_bf16(v16bf a, v16bf b, v8f c) {
    return __builtin_amdgcn_wmma_f32_16x16x32_bf16(
        false, a, false, b, (short)0, c, false, false);
}

// ---------------------------------------------------------------------------
// CDNA5 async global->LDS copy (GVS mode: saddr base + 32-bit voffset).
// Tracked by ASYNCcnt; loads complete in order.
// ---------------------------------------------------------------------------
__device__ __forceinline__ void async_ld16(unsigned lds_byte, const void* base,
                                           unsigned gbyte) {
    asm volatile("global_load_async_to_lds_b128 %0, %1, %2"
                 :: "v"(lds_byte), "v"(gbyte), "s"(base)
                 : "memory");
}
__device__ __forceinline__ void wait_async0() {
    asm volatile("s_wait_asynccnt 0x0" ::: "memory");
}
__device__ __forceinline__ void wait_async6() {
    asm volatile("s_wait_asynccnt 0x6" ::: "memory");
}

// ---------------------------------------------------------------------------
// Prep: fp32 -> bf16 conversion of hidden states
// ---------------------------------------------------------------------------
__global__ void qa_cvt_bf16(const float* __restrict__ x,
                            __bf16* __restrict__ y, int n) {
    int i = blockIdx.x * blockDim.x + threadIdx.x;
    if (i < n) y[i] = to_bf16(x[i]);
}

// ---------------------------------------------------------------------------
// Prep: group-64 dequant  (qw - zero) * scale  -> bf16 weights [O, K] row-major
// ---------------------------------------------------------------------------
__global__ void qa_dequant(const int* __restrict__ qw,
                           const float* __restrict__ sc,
                           const float* __restrict__ zp,
                           __bf16* __restrict__ w) {
    int i = blockIdx.x * blockDim.x + threadIdx.x;   // over 1024*1024
    int o = i >> 10, c = i & 1023, g = c >> 6;
    float s = sc[o * (EMBED / 64) + g];
    float z = zp[o * (EMBED / 64) + g];
    w[i] = to_bf16(((float)qw[i] - z) * s);
}

// ---------------------------------------------------------------------------
// Stage one 128x64 A tile + 64x64 B tile (bf16) into LDS via async copies.
// 256 threads: 6 x b128 async loads per thread = 6 async instrs per wave.
// ---------------------------------------------------------------------------
__device__ __forceinline__ void qa_stage(const __bf16* X, const __bf16* W,
                                         unsigned ldsA, unsigned ldsB,
                                         int m0, int n0, int k0, int tid) {
#pragma unroll
    for (int j = 0; j < 4; ++j) {            // A: 1024 chunks of 16B
        int c = tid + j * 256;
        int r = c >> 3, s = (c & 7) << 3;
        async_ld16(ldsA + (unsigned)((r * 64 + s) * 2), X,
                   (unsigned)(((m0 + r) * EMBED + k0 + s) * 2));
    }
#pragma unroll
    for (int j = 0; j < 2; ++j) {            // B: 512 chunks of 16B
        int c = tid + j * 256;
        int r = c >> 3, s = (c & 7) << 3;
        async_ld16(ldsB + (unsigned)((r * 64 + s) * 2), W,
                   (unsigned)(((n0 + r) * EMBED + k0 + s) * 2));
    }
}

// ---------------------------------------------------------------------------
// Tiled WMMA GEMM:  Y[M=4096, N=1024] = X[4096,1024] * W^T  (+ bias)
//   X bf16 row-major [M,K], W bf16 row-major [N,K] (W rows feed B directly).
//   Workgroup: 256 threads (8 waves), tile 128(M) x 64(N), K step 64,
//   double-buffered LDS with async global->LDS prefetch.
//   epi 0: bf16 out, [B,H,S,D] layout   (Q, K projections)
//   epi 1: bf16 out, [B,H,D,S] layout   (V projection, pre-transposed)
//   epi 2: fp32 out, row-major [M,N]    (O projection -> d_out, NT store)
// ---------------------------------------------------------------------------
__global__ __launch_bounds__(256)
void qa_gemm(const __bf16* __restrict__ X, const __bf16* __restrict__ W,
             const float* __restrict__ bias, void* __restrict__ dst, int epi) {
    __shared__ __bf16 sA[2][128 * 64];       // 2 x 16 KB
    __shared__ __bf16 sB[2][64 * 64];        // 2 x  8 KB

    const int tid  = threadIdx.x;
    const int wave = tid >> 5;
    const int lane = tid & 31;
    const int hi   = (lane & 16) ? 8 : 0;
    const int mn   = lane & 15;
    const int n0   = blockIdx.x * 64;
    const int m0   = blockIdx.y * 128;

    const unsigned ldsA0 = (unsigned)(size_t)&sA[0][0];
    const unsigned ldsA1 = (unsigned)(size_t)&sA[1][0];
    const unsigned ldsB0 = (unsigned)(size_t)&sB[0][0];
    const unsigned ldsB1 = (unsigned)(size_t)&sB[1][0];

    v8f acc[4] = {};

    // Prologue: stage first tile into buffer 0.
    qa_stage(X, W, ldsA0, ldsB0, m0, n0, 0, tid);

    int buf = 0;
    for (int k0 = 0; k0 < EMBED; k0 += 64) {
        if (k0 + 64 < EMBED) {
            // Prefetch next tile into the other buffer, then wait so that
            // only the 6 just-issued loads remain outstanding => current
            // buffer's loads (issued earlier, in-order completion) are done.
            qa_stage(X, W, buf ? ldsA0 : ldsA1, buf ? ldsB0 : ldsB1,
                     m0, n0, k0 + 64, tid);
            wait_async6();
        } else {
            wait_async0();
        }
        __syncthreads();

        const __bf16* cA = sA[buf];
        const __bf16* cB = sB[buf];
#pragma unroll
        for (int kk = 0; kk < 64; kk += 32) {
            v16bf a = load_frag32(&cA[(wave * 16 + mn) * 64 + kk], hi);
#pragma unroll
            for (int t = 0; t < 4; ++t) {
                v16bf b = load_frag32(&cB[(t * 16 + mn) * 64 + kk], hi);
                acc[t] = wmma_bf16(a, b, acc[t]);
            }
        }
        __syncthreads();   // all waves done reading buf before it is refilled
        buf ^= 1;
    }

    // Epilogue: C/D layout -> row = r + 8*(lane>=16), col = lane&15.
    const int rowhi = (lane & 16) ? 8 : 0;
#pragma unroll
    for (int t = 0; t < 4; ++t) {
        const int C = n0 + t * 16 + mn;
        const float bv = bias[C];
#pragma unroll
        for (int r = 0; r < 8; ++r) {
            const int R = m0 + wave * 16 + r + rowhi;
            const float v = acc[t][r] + bv;
            if (epi == 2) {
                __builtin_nontemporal_store(v, &((float*)dst)[(size_t)R * EMBED + C]);
            } else {
                const int b = R >> 11, s = R & (SEQ - 1);
                const int h = C >> 6,  d = C & (HDIM - 1);
                __bf16* D = (__bf16*)dst;
                if (epi == 0)
                    D[(((size_t)(b * HEADS + h)) * SEQ + s) * HDIM + d] = to_bf16(v);
                else
                    D[(((size_t)(b * HEADS + h)) * HDIM + d) * SEQ + s] = to_bf16(v);
            }
        }
    }
}

// ---------------------------------------------------------------------------
// Attention: one workgroup (4 waves) per (b,h, 16 query rows).
//   Phase 1: scores = Q K^T / 8 via WMMA -> fp32 strip [16,2048] in LDS
//   Phase 2: softmax; write P fp32 to d_out (attn_weights, NT) + bf16 to LDS
//   Phase 3: attn_out = P V via WMMA (V pre-transposed [B,H,D,S])
// Dynamic LDS: 16*2048*4 + 16*2048*2 + 2*128*4 = 197632 B (<320 KB WGP LDS)
// ---------------------------------------------------------------------------
extern __shared__ char qa_smem[];

__global__ __launch_bounds__(128)
void qa_attn(const __bf16* __restrict__ Qg, const __bf16* __restrict__ Kg,
             const __bf16* __restrict__ Vtg, __bf16* __restrict__ AO,
             float* __restrict__ AW) {
    float*  sS    = (float*)qa_smem;                            // 16 x 2048 fp32
    __bf16* sP    = (__bf16*)(qa_smem + 16 * SEQ * 4);          // 16 x 2048 bf16
    float*  sRmax = (float*)(qa_smem + 16 * SEQ * 4 + 16 * SEQ * 2);
    float*  sRsum = sRmax + 128;

    const int tid   = threadIdx.x;
    const int wave  = tid >> 5;
    const int lane  = tid & 31;
    const int hi    = (lane & 16) ? 8 : 0;
    const int mn    = lane & 15;
    const int rowhi = (lane & 16) ? 8 : 0;
    const int s0    = blockIdx.x * 16;
    const int bh    = blockIdx.y;              // b*HEADS + h
    const int b     = bh >> 4, h = bh & 15;

    const __bf16* Q  = Qg  + (size_t)bh * SEQ * HDIM;   // rows s, stride 64
    const __bf16* K  = Kg  + (size_t)bh * SEQ * HDIM;   // rows t, stride 64
    const __bf16* Vt = Vtg + (size_t)bh * HDIM * SEQ;   // rows d, stride 2048

    // ---- Phase 1: scores ----
    v16bf a0 = load_frag32(Q + (size_t)(s0 + mn) * HDIM + 0,  hi);
    v16bf a1 = load_frag32(Q + (size_t)(s0 + mn) * HDIM + 32, hi);
    for (int t0 = wave * 512; t0 < wave * 512 + 512; t0 += 16) {
        v8f acc = {};
        v16bf b0 = load_frag32(K + (size_t)(t0 + mn) * HDIM + 0,  hi);
        acc = wmma_bf16(a0, b0, acc);
        v16bf b1 = load_frag32(K + (size_t)(t0 + mn) * HDIM + 32, hi);
        acc = wmma_bf16(a1, b1, acc);
#pragma unroll
        for (int r = 0; r < 8; ++r)
            sS[(r + rowhi) * SEQ + t0 + mn] = acc[r] * 0.125f;  // 1/sqrt(64)
    }
    __syncthreads();

    // ---- Phase 2: softmax (8 threads per row, 256 cols each) ----
    const int row = tid >> 3, part = tid & 7, c0 = part * 256;
    float m = -3.4e38f;
    for (int c = c0; c < c0 + 256; ++c) m = fmaxf(m, sS[row * SEQ + c]);
    sRmax[row * 8 + part] = m;
    __syncthreads();
    m = sRmax[row * 8];
#pragma unroll
    for (int j = 1; j < 8; ++j) m = fmaxf(m, sRmax[row * 8 + j]);
    float sum = 0.f;
    for (int c = c0; c < c0 + 256; ++c) sum += __expf(sS[row * SEQ + c] - m);
    sRsum[row * 8 + part] = sum;
    __syncthreads();
    sum = 0.f;
#pragma unroll
    for (int j = 0; j < 8; ++j) sum += sRsum[row * 8 + j];
    const float inv = 1.0f / sum;
    float* AWrow = AW + ((size_t)bh * SEQ + s0 + row) * SEQ;
    for (int c = c0; c < c0 + 256; ++c) {
        float p = __expf(sS[row * SEQ + c] - m) * inv;
        sP[row * SEQ + c] = to_bf16(p);
        __builtin_nontemporal_store(p, &AWrow[c]);   // 537 MB stream-out, NT
    }
    __syncthreads();

    // ---- Phase 3: attn_out = P @ V, each wave one 16x16 (d) tile ----
    const int d0 = wave * 16;
    v8f acc = {};
    for (int kt = 0; kt < SEQ; kt += 32) {
        v16bf a = load_frag32(&sP[mn * SEQ + kt], hi);
        v16bf bfr = load_frag32(Vt + (size_t)(d0 + mn) * SEQ + kt, hi);
        acc = wmma_bf16(a, bfr, acc);
    }
#pragma unroll
    for (int r = 0; r < 8; ++r) {
        const int s = s0 + r + rowhi;
        const int d = d0 + mn;
        // [B,S,E] layout feeding the O-projection GEMM
        AO[((size_t)(b * SEQ + s)) * EMBED + h * HDIM + d] = to_bf16(acc[r]);
    }
}

// ---------------------------------------------------------------------------
// Launch
// ---------------------------------------------------------------------------
extern "C" void kernel_launch(void* const* d_in, const int* in_sizes, int n_in,
                              void* d_out, int out_size, void* d_ws, size_t ws_size,
                              hipStream_t stream) {
    (void)in_sizes; (void)n_in; (void)out_size; (void)ws_size;

    const float* hidden = (const float*)d_in[0];
    const int*   qw[4]; const float* sc[4]; const float* zp[4]; const float* bs[4];
    for (int p = 0; p < 4; ++p) {
        qw[p] = (const int*)  d_in[1 + p * 4];
        sc[p] = (const float*)d_in[2 + p * 4];
        zp[p] = (const float*)d_in[3 + p * 4];
        bs[p] = (const float*)d_in[4 + p * 4];
    }

    // Workspace layout (bf16 buffers), 48 MB total.
    char* ws = (char*)d_ws;
    const size_t MB = 1024u * 1024u;
    __bf16* Xb  = (__bf16*)(ws + 0 * MB);       // [4096,1024]       8 MB
    __bf16* Wb0 = (__bf16*)(ws + 8 * MB);       // q weights         2 MB
    __bf16* Wb1 = (__bf16*)(ws + 10 * MB);      // k weights         2 MB
    __bf16* Wb2 = (__bf16*)(ws + 12 * MB);      // v weights         2 MB
    __bf16* Wb3 = (__bf16*)(ws + 14 * MB);      // o weights         2 MB
    __bf16* Qb  = (__bf16*)(ws + 16 * MB);      // [B,H,S,D]         8 MB
    __bf16* Kb  = (__bf16*)(ws + 24 * MB);      // [B,H,S,D]         8 MB
    __bf16* Vtb = (__bf16*)(ws + 32 * MB);      // [B,H,D,S]         8 MB
    __bf16* AOb = (__bf16*)(ws + 40 * MB);      // [B,S,E]           8 MB
    __bf16* Wb[4] = {Wb0, Wb1, Wb2, Wb3};

    float* out_main = (float*)d_out;                         // [B,S,E]
    float* out_attn = (float*)d_out + (size_t)MROWS * EMBED; // [B,H,S,S]

    // 1) Prep
    qa_cvt_bf16<<<(MROWS * EMBED) / 256, 256, 0, stream>>>(hidden, Xb, MROWS * EMBED);
    for (int p = 0; p < 4; ++p)
        qa_dequant<<<(EMBED * EMBED) / 256, 256, 0, stream>>>(qw[p], sc[p], zp[p], Wb[p]);

    // 2) Q/K/V projections (WMMA, async double-buffered staging)
    dim3 gg(EMBED / 64, MROWS / 128);
    qa_gemm<<<gg, 256, 0, stream>>>(Xb, Wb[0], bs[0], (void*)Qb,  0);
    qa_gemm<<<gg, 256, 0, stream>>>(Xb, Wb[1], bs[1], (void*)Kb,  0);
    qa_gemm<<<gg, 256, 0, stream>>>(Xb, Wb[2], bs[2], (void*)Vtb, 1);

    // 3) Attention (WMMA + softmax), 197632 B dynamic LDS per workgroup
    size_t smem = (size_t)16 * SEQ * 4 + (size_t)16 * SEQ * 2 + 2 * 128 * 4;
    dim3 ga(SEQ / 16, BATCH * HEADS);
    qa_attn<<<ga, 128, smem, stream>>>(Qb, Kb, Vtb, AOb, out_attn);

    // 4) Output projection (WMMA, fp32 NT epilogue straight to d_out)
    qa_gemm<<<gg, 256, 0, stream>>>(AOb, Wb[3], bs[3], (void*)out_main, 2);
}
